// LightningIndexer_56358560858474
// MI455X (gfx1250) — matrix operand order repaired and checked by reading, not compiled, not run on hardware
//
#include <hip/hip_runtime.h>
#include <hip/hip_bf16.h>

// ---------------------------------------------------------------------------
// LightningIndexer for MI455X (gfx1250, wave32, WMMA bf16, 320KB LDS/WGP, TDM)
//
// Pipeline:
//   cast x,Wqd,Wqu,Wk,Ww -> bf16
//   GEMM1: qd  = x  @ Wqd   (4096 x 1792), bf16 out      \
//   GEMM2: q   = qd @ Wqu   (4096 x 4096), bf16 out       | A-tile staged via
//   GEMM3: km  = x  @ Wk    (4096 x 128),  bf16 out       | TDM tensor_load_to_lds
//   GEMM4: wf  = x  @ Ww    (4096 x 32),   fp32 out      /  (padded LDS rows)
//   score_topk: 16 query rows x 4096 keys of fp32 scores resident in LDS
//               (256KB of the 320KB WGP LDS), WMMA bf16 QK^T fused with
//               relu/head-weighting + causal mask + 4-pass radix top-2048.
// ---------------------------------------------------------------------------

#define T_DIM   4096
#define D_DIM   7168
#define DQ_DIM  1792
#define H_DIM   32
#define DH_DIM  128
#define HD_DIM  4096   // H*DH
#define TOPK_K  2048

typedef __attribute__((ext_vector_type(16))) __bf16 v16bf;
typedef __attribute__((ext_vector_type(8)))  float  v8f;
typedef __attribute__((ext_vector_type(4)))  unsigned int v4u;
typedef __attribute__((ext_vector_type(8)))  int    v8i;
typedef __attribute__((ext_vector_type(4)))  int    v4i;

#if defined(__gfx1250__) && __has_builtin(__builtin_amdgcn_tensor_load_to_lds)
#define USE_TDM 1
#else
#define USE_TDM 0
#endif

union FragBF {
    v16bf v;
    uint4 q[2];
};

__device__ __forceinline__ v8f v8f_zero() {
    v8f z;
#pragma unroll
    for (int i = 0; i < 8; ++i) z[i] = 0.0f;
    return z;
}

__device__ __forceinline__ unsigned short f32_to_bf16_rne(float f) {
    unsigned int u = __float_as_uint(f);
    unsigned int r = (u + 0x7FFFu + ((u >> 16) & 1u)) >> 16;
    return (unsigned short)r;
}

// monotone key: larger float -> larger unsigned
__device__ __forceinline__ unsigned int fkey(float f) {
    unsigned int u = __float_as_uint(f);
    return (u & 0x80000000u) ? ~u : (u | 0x80000000u);
}

// ---------------------------------------------------------------------------
// TDM: load a (rows x 32) bf16 tile from a row-major (Mrows x Kels) tensor
// into LDS at byte offset lds_off, with 16B padding after each 64B row
// (-> 80B LDS row stride == LDS_AS bf16 elements).
//   D# group0: count=1 | lds_addr | global_addr | type=2
//   D# group1: data_size=2B, pad_enable, pad_interval=16 DW, pad_amount=4 DW,
//              tensor_dim0=Kels, tensor_dim1=Mrows, tile_dim0=32, tile_dim1=rows,
//              tensor_dim0_stride=Kels
// One TDM op per wave (EXEC ignored); completion via TENSORcnt.
// ---------------------------------------------------------------------------
__device__ __forceinline__ void tdm_load_a_tile(const unsigned short* gsrc,
                                                unsigned int lds_off, int Kels,
                                                int Mrows, int rows) {
#if USE_TDM
    unsigned long long ga = (unsigned long long)gsrc;
    v4u g0;
    g0[0] = 1u;                                          // count=1, user D#
    g0[1] = lds_off;                                     // lds_addr (bytes)
    g0[2] = (unsigned int)(ga & 0xFFFFFFFFull);          // global_addr[31:0]
    g0[3] = (unsigned int)((ga >> 32) & 0x1FFFFFFull)    // global_addr[56:32]
          | (2u << 30);                                  // type = 2 ("image")
    v8i g1;
    g1[0] = (int)((1u << 16)      // data_size = 1 -> 2 bytes
          | (1u << 20)            // pad_enable
          | (3u << 22)            // pad_interval code 3 -> every 16 DWORDs
          | (3u << 25));          // pad_amount   code 3 -> 4 DWORDs (16B)
    g1[1] = (int)(((unsigned)Kels & 0xFFFFu) << 16);     // tensor_dim0[15:0]
    g1[2] = (int)((((unsigned)Kels >> 16) & 0xFFFFu)     // tensor_dim0[31:16]
          | (((unsigned)Mrows & 0xFFFFu) << 16));        // tensor_dim1[15:0]
    g1[3] = (int)((((unsigned)Mrows >> 16) & 0xFFFFu)    // tensor_dim1[31:16]
          | (32u << 16));                                // tile_dim0 = 32
    g1[4] = (int)((unsigned)rows & 0xFFFFu);             // tile_dim1 (tile_dim2=0)
    g1[5] = (int)Kels;                                   // tensor_dim0_stride[31:0]
    g1[6] = 0;                                           // stride[47:32] | dim1_stride lo
    g1[7] = 0;
    v4i z4;
    z4[0] = z4[1] = z4[2] = z4[3] = 0;
#if __clang_major__ >= 23
    v8i z8;
#pragma unroll
    for (int i = 0; i < 8; ++i) z8[i] = 0;
    __builtin_amdgcn_tensor_load_to_lds(g0, g1, z4, z4, z8, 0);
#else
    __builtin_amdgcn_tensor_load_to_lds(g0, g1, z4, z4, 0);
#endif
#else
    (void)gsrc; (void)lds_off; (void)Kels; (void)Mrows; (void)rows;
#endif
}

// ---------------------------------------------------------------------------
// fp32 -> bf16 cast (grid-stride)
// ---------------------------------------------------------------------------
__global__ __launch_bounds__(256) void cast_f32_bf16_kernel(
    const float* __restrict__ in, unsigned short* __restrict__ out, long n) {
    long i = (long)blockIdx.x * blockDim.x + threadIdx.x;
    long stride = (long)gridDim.x * blockDim.x;
    for (; i < n; i += stride) out[i] = f32_to_bf16_rne(in[i]);
}

// ---------------------------------------------------------------------------
// Generic bf16 WMMA GEMM: C(M,N) = A(M,K) @ B(K,N); A,B row-major bf16.
// Block tile 128x64, 8 waves, each wave a 32x32 subtile (2x2 WMMA 16x16),
// K stepped in chunks of 32. A tile staged by TDM (1/8 tile per wave, padded
// LDS rows); B tile staged manually transposed (TDM has no transpose).
// Requires M % 128 == 0 and K % 32 == 0 (true here); N guarded.
// ---------------------------------------------------------------------------
#define GBM 128
#define GBN 64
#define GKB 32
#define LDS_AS 40   // bf16 elems per LDS A row (80B stride; matches TDM pad)
#define LDS_BS 40   // bf16 elems per LDS Bt row

template <int OUT_BF16>
__global__ __launch_bounds__(256) void gemm_bf16_wmma_kernel(
    const unsigned short* __restrict__ A, const unsigned short* __restrict__ B,
    void* __restrict__ Cout, int M, int N, int K) {
    __shared__ unsigned short As[GBM * LDS_AS];
    __shared__ unsigned short Bt[GBN * LDS_BS];   // transposed: Bt[n][k]

    const int tid  = threadIdx.x;
    const int lane = tid & 31;
    const int wave = tid >> 5;
    const int wm   = wave >> 1;     // 0..3  (M direction)
    const int wn   = wave & 1;      // 0..1  (N direction)
    const int l15  = lane & 15;
    const int kh   = lane >> 4;     // K-half selector for fragments
    const int m0   = blockIdx.y * GBM;
    const int n0   = blockIdx.x * GBN;

    const unsigned int as_base =
        (unsigned int)(unsigned long long)(void*)&As[0];

    v8f acc[2][2];
#pragma unroll
    for (int r = 0; r < 2; ++r)
#pragma unroll
        for (int c = 0; c < 2; ++c) acc[r][c] = v8f_zero();

    for (int k0 = 0; k0 < K; k0 += GKB) {
#if USE_TDM
        // --- A tile via Tensor Data Mover: each wave DMAs 16 rows x 32 cols ---
        tdm_load_a_tile(A + (long)(m0 + wave * 16) * K + k0,
                        as_base + (unsigned)(wave * 16 * LDS_AS * 2), K, M, 16);
#else
        // --- fallback: manual A staging (host-pass parse / no-TDM toolchain) ---
#pragma unroll
        for (int i = 0; i < 2; ++i) {
            int linear = tid + i * 256;           // 0..511
            int row = linear >> 2;                // 0..127
            int col = (linear & 3) * 8;           // 0,8,16,24
            uint4 d = *reinterpret_cast<const uint4*>(
                A + (long)(m0 + row) * K + k0 + col);
            *reinterpret_cast<uint4*>(&As[row * LDS_AS + col]) = d;
        }
#endif
        // --- B tile 32x64 staged transposed into Bt[n][k] (guard N) ---
        {
            int krow = tid >> 3;                  // 0..31
            int col  = (tid & 7) * 8;             // 0..56
            const unsigned short* src = B + (long)(k0 + krow) * N + n0 + col;
#pragma unroll
            for (int j = 0; j < 8; ++j) {
                unsigned short v =
                    (n0 + col + j < N) ? src[j] : (unsigned short)0;
                Bt[(col + j) * LDS_BS + krow] = v;
            }
        }
        // prefetch next A chunk into GL2 while we compute
        if (k0 + GKB < K)
            __builtin_prefetch(A + (long)(m0 + (tid >> 1)) * K + k0 + GKB, 0, 1);
#if USE_TDM
        __builtin_amdgcn_s_wait_tensorcnt(0);     // our wave's TDM slice done
#endif
        __syncthreads();                          // publish all slices + Bt

        FragBF a[2], b[2];
#pragma unroll
        for (int r = 0; r < 2; ++r) {
            const unsigned short* ap = &As[(wm * 32 + r * 16 + l15) * LDS_AS];
            a[r].q[0] = *reinterpret_cast<const uint4*>(ap + kh * 8);
            a[r].q[1] = *reinterpret_cast<const uint4*>(ap + 16 + kh * 8);
            const unsigned short* bp =
                &Bt[(wn * 32 + r * 16 + l15) * LDS_BS + kh * 8];
            b[r].q[0] = *reinterpret_cast<const uint4*>(bp);
            b[r].q[1] = *reinterpret_cast<const uint4*>(bp + 16);
        }
#pragma unroll
        for (int r = 0; r < 2; ++r)
#pragma unroll
            for (int c = 0; c < 2; ++c)
                acc[r][c] = __builtin_amdgcn_wmma_f32_16x16x32_bf16(
                    false, a[r].v, false, b[c].v, (short)0, acc[r][c], false,
                    false);
        __syncthreads();
    }

    // store C (fp32 C/D layout: VGPR r -> M = r or r+8; lane&15 -> N)
#pragma unroll
    for (int r = 0; r < 2; ++r)
#pragma unroll
        for (int c = 0; c < 2; ++c) {
            int n = n0 + wn * 32 + c * 16 + l15;
            if (n >= N) continue;
#pragma unroll
            for (int vr = 0; vr < 8; ++vr) {
                int m = m0 + wm * 32 + r * 16 + vr + kh * 8;
                float val = acc[r][c][vr];
                if (OUT_BF16) {
                    reinterpret_cast<unsigned short*>(Cout)[(long)m * N + n] =
                        f32_to_bf16_rne(val);
                } else {
                    reinterpret_cast<float*>(Cout)[(long)m * N + n] = val;
                }
            }
        }
}

// ---------------------------------------------------------------------------
// Fused score + causal mask + top-2048 per query block of 16.
// Dynamic LDS: 16 x 4096 fp32 score rows = 256 KB (fits in 320KB WGP LDS).
// 8 waves: key blocks striped over waves; per (key block, head):
// 4 chained v_wmma_f32_16x16x32_bf16 over DH=128, then relu * w[t,h] reduce.
// Selection: 4-pass 8-bit radix select (MSB first) per row via LDS histograms.
// ---------------------------------------------------------------------------
__global__ __launch_bounds__(256) void score_topk_kernel(
    const unsigned short* __restrict__ qmat,   // (T, H*DH) bf16
    const unsigned short* __restrict__ kmat,   // (T, DH)   bf16
    const float* __restrict__ wf,              // (T, H)    fp32
    int* __restrict__ out)                     // (T, TOPK) int32
{
    extern __shared__ unsigned char smem_raw[];
    float* sc = reinterpret_cast<float*>(smem_raw);   // [16][4096]

    __shared__ unsigned int hist[16 * 256];
    __shared__ float        wsh[16 * 32];
    __shared__ unsigned int prefix[16];
    __shared__ unsigned int kneed[16];
    __shared__ unsigned int cnt[16];

    const int tid  = threadIdx.x;
    const int lane = tid & 31;
    const int wave = tid >> 5;
    const int l15  = lane & 15;
    const int kh   = lane >> 4;
    const int tb   = blockIdx.x;
    const int t0   = tb * 16;

    // ---- init: -inf score rows, stage w block, reset select state ----
    for (int i = tid; i < 16 * T_DIM; i += 256) sc[i] = -__builtin_inff();
    for (int i = tid; i < 16 * H_DIM; i += 256) wsh[i] = wf[(long)t0 * H_DIM + i];
    if (tid < 16) {
        prefix[tid] = 0u;
        kneed[tid]  = TOPK_K;
        cnt[tid]    = 0u;
    }
    __syncthreads();

    // ---- score compute: key blocks striped over the 8 waves ----
    const int nkb = tb + 1;   // causal: only key blocks s0 <= t0+15
    for (int kb = wave; kb < nkb; kb += 8) {
        const int s0 = kb * 16;
        // B fragments: lane l15 = key column; dh {8kh..8kh+7, 16+8kh..16+8kh+7} per lane ->
        // two 16B global loads (k row-major (T,DH)).
        FragBF bfr[4];
#pragma unroll
        for (int c = 0; c < 4; ++c) {
            const unsigned short* bp =
                kmat + (long)(s0 + l15) * DH_DIM + c * 32 + kh * 8;
            bfr[c].q[0] = *reinterpret_cast<const uint4*>(bp);
            bfr[c].q[1] = *reinterpret_cast<const uint4*>(bp + 16);
        }

        v8f outacc = v8f_zero();
        for (int h = 0; h < H_DIM; ++h) {
            const unsigned short* qrow =
                qmat + (long)(t0 + l15) * HD_DIM + h * DH_DIM;
            v8f acc = v8f_zero();
#pragma unroll
            for (int c = 0; c < 4; ++c) {
                FragBF a;
                a.q[0] = *reinterpret_cast<const uint4*>(qrow + c * 32 + kh * 8);
                a.q[1] = *reinterpret_cast<const uint4*>(qrow + c * 32 + 16 + kh * 8);
                acc = __builtin_amdgcn_wmma_f32_16x16x32_bf16(
                    false, a.v, false, bfr[c].v, (short)0, acc, false, false);
            }
            // relu + per-(t,h) weight, accumulate over heads
#pragma unroll
            for (int vr = 0; vr < 8; ++vr) {
                int row = vr + kh * 8;             // local query row 0..15
                outacc[vr] += fmaxf(acc[vr], 0.0f) * wsh[row * H_DIM + h];
            }
        }

        // causal mask + write to LDS score rows
        const int s = s0 + l15;
#pragma unroll
        for (int vr = 0; vr < 8; ++vr) {
            int row = vr + kh * 8;
            int t = t0 + row;
            sc[row * T_DIM + s] = (s <= t) ? outacc[vr] : -__builtin_inff();
        }
    }
    __syncthreads();

    // ---- top-k: each wave selects for rows {wave, wave+8} ----
    const int rows0 = wave;
    const int rows1 = wave + 8;

    for (int pass = 0; pass < 4; ++pass) {
        const int shift = 24 - pass * 8;
        for (int i = tid; i < 16 * 256; i += 256) hist[i] = 0u;
        __syncthreads();

#pragma unroll
        for (int rr = 0; rr < 2; ++rr) {
            const int row = rr ? rows1 : rows0;
            const unsigned int pfx = prefix[row];
            const unsigned int hmask =
                (pass == 0) ? 0u : (0xFFFFFFFFu << (shift + 8));
            unsigned int* hrow = &hist[row * 256];
            for (int j = lane; j < T_DIM; j += 32) {
                unsigned int u = fkey(sc[row * T_DIM + j]);
                if ((u & hmask) == pfx)
                    atomicAdd(&hrow[(u >> shift) & 255u], 1u);
            }
        }
        __syncthreads();

        if (lane == 0) {
#pragma unroll
            for (int rr = 0; rr < 2; ++rr) {
                const int row = rr ? rows1 : rows0;
                unsigned int need = kneed[row];
                unsigned int* hrow = &hist[row * 256];
                int b = 255;
                for (; b > 0; --b) {
                    unsigned int c = hrow[b];
                    if (c >= need) break;
                    need -= c;
                }
                kneed[row] = need;
                prefix[row] |= ((unsigned int)b) << shift;
            }
        }
        __syncthreads();
    }

    // emit: strictly-greater first, then threshold-equal fill to TOPK
#pragma unroll
    for (int rr = 0; rr < 2; ++rr) {
        const int row = rr ? rows1 : rows0;
        const unsigned int thr = prefix[row];
        int* orow = out + (long)(t0 + row) * TOPK_K;
        for (int j = lane; j < T_DIM; j += 32) {
            unsigned int u = fkey(sc[row * T_DIM + j]);
            if (u > thr) {
                unsigned int pos = atomicAdd(&cnt[row], 1u);
                orow[pos] = j;
            }
        }
    }
    __syncthreads();
#pragma unroll
    for (int rr = 0; rr < 2; ++rr) {
        const int row = rr ? rows1 : rows0;
        const unsigned int thr = prefix[row];
        int* orow = out + (long)(t0 + row) * TOPK_K;
        for (int j = lane; j < T_DIM; j += 32) {
            unsigned int u = fkey(sc[row * T_DIM + j]);
            if (u == thr) {
                unsigned int pos = atomicAdd(&cnt[row], 1u);
                if (pos < TOPK_K) orow[pos] = j;
            }
        }
    }
}

// ---------------------------------------------------------------------------
// host launch
// ---------------------------------------------------------------------------
extern "C" void kernel_launch(void* const* d_in, const int* in_sizes, int n_in,
                              void* d_out, int out_size, void* d_ws,
                              size_t ws_size, hipStream_t stream) {
    (void)in_sizes; (void)n_in; (void)out_size; (void)ws_size;
    const float* x   = (const float*)d_in[0];   // (1, 4096, 7168)
    const float* Wqd = (const float*)d_in[1];   // (7168, 1792)
    const float* Wqu = (const float*)d_in[2];   // (1792, 4096)
    const float* Wk  = (const float*)d_in[3];   // (7168, 128)
    const float* Ww  = (const float*)d_in[4];   // (7168, 32)
    int* out = (int*)d_out;                     // (1, 4096, 2048) int32

    // scratch layout in d_ws (~151 MB total)
    char* ws = (char*)d_ws;
    size_t off = 0;
    auto alloc = [&](size_t bytes) -> void* {
        void* p = (void*)(ws + off);
        off += (bytes + 255) & ~(size_t)255;
        return p;
    };
    unsigned short* xb   = (unsigned short*)alloc((size_t)T_DIM * D_DIM * 2);
    unsigned short* wqdb = (unsigned short*)alloc((size_t)D_DIM * DQ_DIM * 2);
    unsigned short* wqub = (unsigned short*)alloc((size_t)DQ_DIM * HD_DIM * 2);
    unsigned short* wkb  = (unsigned short*)alloc((size_t)D_DIM * DH_DIM * 2);
    unsigned short* wwb  = (unsigned short*)alloc((size_t)D_DIM * H_DIM * 2);
    unsigned short* qdb  = (unsigned short*)alloc((size_t)T_DIM * DQ_DIM * 2);
    unsigned short* qb   = (unsigned short*)alloc((size_t)T_DIM * HD_DIM * 2);
    unsigned short* kbuf = (unsigned short*)alloc((size_t)T_DIM * DH_DIM * 2);
    float*          wfbf = (float*)alloc((size_t)T_DIM * H_DIM * 4);

    // casts
    cast_f32_bf16_kernel<<<4096, 256, 0, stream>>>(x, xb, (long)T_DIM * D_DIM);
    cast_f32_bf16_kernel<<<4096, 256, 0, stream>>>(Wqd, wqdb, (long)D_DIM * DQ_DIM);
    cast_f32_bf16_kernel<<<2048, 256, 0, stream>>>(Wqu, wqub, (long)DQ_DIM * HD_DIM);
    cast_f32_bf16_kernel<<<512, 256, 0, stream>>>(Wk, wkb, (long)D_DIM * DH_DIM);
    cast_f32_bf16_kernel<<<256, 256, 0, stream>>>(Ww, wwb, (long)D_DIM * H_DIM);

    // GEMM1: qd = x @ Wqd   (4096 x 1792, K=7168)
    gemm_bf16_wmma_kernel<1><<<dim3(DQ_DIM / GBN, T_DIM / GBM), 256, 0, stream>>>(
        xb, wqdb, qdb, T_DIM, DQ_DIM, D_DIM);
    // GEMM2: q = qd @ Wqu   (4096 x 4096, K=1792)
    gemm_bf16_wmma_kernel<1><<<dim3(HD_DIM / GBN, T_DIM / GBM), 256, 0, stream>>>(
        qdb, wqub, qb, T_DIM, HD_DIM, DQ_DIM);
    // GEMM3: k = x @ Wk     (4096 x 128, K=7168)
    gemm_bf16_wmma_kernel<1><<<dim3(DH_DIM / GBN, T_DIM / GBM), 256, 0, stream>>>(
        xb, wkb, kbuf, T_DIM, DH_DIM, D_DIM);
    // GEMM4: w = x @ Ww     (4096 x 32, K=7168), fp32 out, N guarded
    gemm_bf16_wmma_kernel<0><<<dim3(1, T_DIM / GBM), 256, 0, stream>>>(
        xb, wwb, wfbf, T_DIM, H_DIM, D_DIM);

    // fused score + causal + top-2048 per 16-query block
    const size_t dyn_lds = (size_t)16 * T_DIM * sizeof(float);   // 256 KB
    hipFuncSetAttribute((const void*)score_topk_kernel,
                        hipFuncAttributeMaxDynamicSharedMemorySize,
                        (int)dyn_lds);
    score_topk_kernel<<<T_DIM / 16, 256, dyn_lds, stream>>>(qb, kbuf, wfbf, out);
}
